// BahdanauAttention_3805341024261
// MI455X (gfx1250) — matrix operand order repaired and checked by reading, not compiled, not run on hardware
//
#include <hip/hip_runtime.h>
#include <math.h>

typedef __attribute__((ext_vector_type(16))) __bf16 v16bf;
typedef __attribute__((ext_vector_type(8)))  __bf16 v8bf;
typedef __attribute__((ext_vector_type(4)))  __bf16 v4bf;
typedef __attribute__((ext_vector_type(8)))  float  v8f;

#define B_DIM   64
#define S_DIM   1000
#define DEC     512
#define K_E     1024              // 2*ENC
#define M_TILE  64                // rows per workgroup (4 x 16-row WMMA tiles)
#define ROWS_TOTAL (B_DIM * S_DIM)   // 64000

// ---------------------------------------------------------------------------
// Kernel 1: WT[n*1024 + k] = bf16( attn_w[(512+k)*512 + n] )   (W_e transposed)
// ---------------------------------------------------------------------------
__global__ void wt_convert_kernel(const float* __restrict__ attn_w,
                                  unsigned short* __restrict__ wt_u) {
    int gid = blockIdx.x * blockDim.x + threadIdx.x;      // 0 .. 512*1024-1
    int n = gid >> 10;
    int k = gid & 1023;
    __bf16 h = (__bf16)attn_w[(size_t)(DEC + k) * DEC + n];
    wt_u[gid] = *(unsigned short*)&h;
}

// ---------------------------------------------------------------------------
// Kernel 2: pre[b,d] = attn_b[d] + sum_k hidden[b,k] * attn_w[k,d]  (f32)
// ---------------------------------------------------------------------------
__global__ void pre_kernel(const float* __restrict__ hidden,
                           const float* __restrict__ attn_w,
                           const float* __restrict__ attn_b,
                           float* __restrict__ pre) {
    __shared__ float h[DEC];
    int b = blockIdx.x;
    int d = threadIdx.x;                  // blockDim.x == 512
    h[d] = hidden[b * DEC + d];
    __syncthreads();
    float acc = attn_b[d];
    for (int k = 0; k < DEC; ++k)
        acc = fmaf(h[k], attn_w[(size_t)k * DEC + d], acc);
    pre[b * DEC + d] = acc;
}

// ---------------------------------------------------------------------------
// Kernel 3: main fused GEMM + tanh + dot(v)  ->  att[64000]
//   grid = 1000, block = 256 (8 waves). WG handles 64 rows x 512 cols x K=1024.
//   wave w owns columns [w*64, w*64+64); all waves cover all 4 M sub-tiles.
//   Dynamic LDS: 64 rows x 1024 K bf16 = 128 KB (reused as f32 scratch after).
// ---------------------------------------------------------------------------
__global__ void __launch_bounds__(256)
attn_energy_kernel(const float* __restrict__ enc,
                   const unsigned short* __restrict__ wt_u,
                   const float* __restrict__ pre,
                   const float* __restrict__ vvec,
                   float* __restrict__ att) {
    extern __shared__ __align__(32) char smem[];
    __bf16* As = (__bf16*)smem;                          // [M_TILE][K_E]
    const __bf16* wt = (const __bf16*)wt_u;

    const int tid   = threadIdx.x;
    const int wv    = tid >> 5;
    const int lane  = tid & 31;
    const int half  = lane >> 4;
    const int l16   = lane & 15;
    const int rowBase = (int)blockIdx.x * M_TILE;

    // ---- stage A: 64 rows x 1024 f32 -> bf16 in LDS (coalesced float4 loads)
    const float4* encV = (const float4*)(enc + (size_t)rowBase * K_E);
    #pragma unroll 4
    for (int i = 0; i < 64; ++i) {
        int fi = tid + i * 256;          // float4 index < 16384
        float4 x = encV[fi];
        v4bf y;
        y[0] = (__bf16)x.x; y[1] = (__bf16)x.y;
        y[2] = (__bf16)x.z; y[3] = (__bf16)x.w;
        *(v4bf*)(&As[fi * 4]) = y;
    }
    __syncthreads();

    // ---- K loop: 32 blocks of K=32, 16 WMMAs each (4 M-tiles x 4 N-tiles)
    v8f c[4][4] = {};
    const int nbase0 = wv * 64;
    for (int kb = 0; kb < 32; ++kb) {
        v16bf a[4];
        #pragma unroll
        for (int mt = 0; mt < 4; ++mt) {
            // A k order used here (same as B): half h -> K[16h..16h+15] contiguous
            const __bf16* ap = &As[(mt * 16 + l16) * K_E + kb * 32 + half * 16];
            union { v16bf v; v8bf h2[2]; } ua;
            ua.h2[0] = *(const v8bf*)ap;
            ua.h2[1] = *(const v8bf*)(ap + 8);
            a[mt] = ua.v;
        }
        #pragma unroll
        for (int nt = 0; nt < 4; ++nt) {
            // B k order used here (same as A): lane = column, K = half*16 .. half*16+15 contiguous
            const __bf16* bp = wt + (size_t)(nbase0 + nt * 16 + l16) * K_E
                               + kb * 32 + half * 16;
            v16bf bfrag = *(const v16bf*)bp;
            #pragma unroll
            for (int mt = 0; mt < 4; ++mt) {
                c[mt][nt] = __builtin_amdgcn_wmma_f32_16x16x32_bf16(
                    false, a[mt], false, bfrag, (short)0, c[mt][nt], false, false);
            }
        }
    }
    __syncthreads();                     // done with As -> reuse as f32 scratch

    // ---- fused epilogue: att_partial = sum_n tanh(c + pre) * v
    float* attLds = (float*)smem;        // [M_TILE]
    if (tid < M_TILE) attLds[tid] = 0.0f;
    __syncthreads();

    #pragma unroll
    for (int mt = 0; mt < 4; ++mt) {
        #pragma unroll
        for (int r = 0; r < 8; ++r) {
            // C layout: lane half selects M = r or r+8; N = lane&15
            int mloc = mt * 16 + r + 8 * half;
            int row  = rowBase + mloc;
            const float* preRow = pre + (row / S_DIM) * DEC;
            float s = 0.0f;
            #pragma unroll
            for (int nt = 0; nt < 4; ++nt) {
                int col = nbase0 + nt * 16 + l16;
                float e = tanhf(c[mt][nt][r] + preRow[col]);
                s = fmaf(e, vvec[col], s);
            }
            // reduce over the 16 lanes of this half-wave
            s += __shfl_xor(s, 1);
            s += __shfl_xor(s, 2);
            s += __shfl_xor(s, 4);
            s += __shfl_xor(s, 8);
            if (l16 == 0) atomicAdd(&attLds[mloc], s);   // ds_add_f32 across waves
        }
    }
    __syncthreads();
    if (tid < M_TILE) att[rowBase + tid] = attLds[tid];
}

// ---------------------------------------------------------------------------
// Kernel 4: row softmax over S=1000 per batch
// ---------------------------------------------------------------------------
__global__ void softmax_kernel(const float* __restrict__ att,
                               float* __restrict__ out) {
    __shared__ float red[256];
    int b = blockIdx.x, tid = threadIdx.x;
    const float* x = att + b * S_DIM;
    float m = -INFINITY;
    for (int i = tid; i < S_DIM; i += 256) m = fmaxf(m, x[i]);
    red[tid] = m; __syncthreads();
    for (int off = 128; off > 0; off >>= 1) {
        if (tid < off) red[tid] = fmaxf(red[tid], red[tid + off]);
        __syncthreads();
    }
    m = red[0]; __syncthreads();
    float s = 0.0f;
    for (int i = tid; i < S_DIM; i += 256) {
        float e = __expf(x[i] - m);
        out[b * S_DIM + i] = e;
        s += e;
    }
    red[tid] = s; __syncthreads();
    for (int off = 128; off > 0; off >>= 1) {
        if (tid < off) red[tid] += red[tid + off];
        __syncthreads();
    }
    float inv = 1.0f / red[0];
    for (int i = tid; i < S_DIM; i += 256) out[b * S_DIM + i] *= inv;
}

// ---------------------------------------------------------------------------
extern "C" void kernel_launch(void* const* d_in, const int* in_sizes, int n_in,
                              void* d_out, int out_size, void* d_ws, size_t ws_size,
                              hipStream_t stream) {
    const float* hidden = (const float*)d_in[0];   // [64, 512]
    const float* enc    = (const float*)d_in[1];   // [64, 1000, 1024]
    const float* attn_w = (const float*)d_in[2];   // [1536, 512]
    const float* attn_b = (const float*)d_in[3];   // [512]
    const float* vvec   = (const float*)d_in[4];   // [512]
    float* out = (float*)d_out;                    // [64, 1000]

    char* ws = (char*)d_ws;
    unsigned short* wt = (unsigned short*)ws;                       // 1 MB bf16 W_e^T
    float* pre = (float*)(ws + (1 << 20));                          // 128 KB
    float* att = (float*)(ws + (1 << 20) + B_DIM * DEC * 4);        // 256 KB

    wt_convert_kernel<<<(DEC * K_E) / 256, 256, 0, stream>>>(attn_w, wt);
    pre_kernel<<<B_DIM, DEC, 0, stream>>>(hidden, attn_w, attn_b, pre);

    size_t ldsBytes = (size_t)M_TILE * K_E * sizeof(__bf16);        // 128 KB dynamic
    attn_energy_kernel<<<ROWS_TOTAL / M_TILE, 256, ldsBytes, stream>>>(
        enc, wt, pre, vvec, att);

    softmax_kernel<<<B_DIM, 256, 0, stream>>>(att, out);
}
